// SpectrumExtractionBySTRAIGHT_69956427317948
// MI455X (gfx1250) — compile-verified, matrix operand order checked by reading, not run on hardware
//
#include <hip/hip_runtime.h>
#include <math.h>

// ---------------- problem constants (match reference) ----------------
#define SRATE   16000
#define FP      80
#define FFTL    2048
#define FL      1280            // SR*80/1000
#define OSL     1025            // FFTL/2+1
#define BATCH   4
#define NFRAME  1500
#define TLEN    (FP*NFRAME)     // 120000
#define NFR     (BATCH*NFRAME)  // 6000
#define F0PM    54
#define F0P2M   27
#define LBB     37              // round(300/SR*FFTL)-1
#define DEFAULT_F0 160.0
#define EPSf    1e-8f
#define FNOMf   40.0f
#define PI_F    3.14159265358979323846f
#define TAU_F   6.28318530717958647692f

#define NPAD    1056            // OSL padded to multiple of 32 (K of hfft GEMMs)
#define NRF     2112            // 2*NPAD (re | im) rfft output columns
#define KT_FFTL 64              // FFTL/32
#define KT_NPAD 33              // NPAD/32
#define NT_RF   132             // NRF/16  (multiple of 4)
#define NT_FFTL 128             // FFTL/16 (multiple of 4)
#define NT_NPAD_AL 68           // NPAD/16=66 padded to multiple of 4

// CDNA5 async global->LDS path (guarded; falls back to vector copy)
#if __has_builtin(__builtin_amdgcn_global_load_async_to_lds_b128) && \
    __has_builtin(__builtin_amdgcn_s_wait_asynccnt)
#define HAVE_ASYNC_LDS 1
#endif

typedef __bf16 bf16_t;
typedef __attribute__((ext_vector_type(16))) __bf16 v16bf;
typedef __attribute__((ext_vector_type(8)))  float  v8f;
typedef int v4i_t __attribute__((vector_size(16)));   // matches builtin param type

// ---------------- WMMA fragment-order swizzles (global memory) --------
// A (MxK): lane = (m&15) + 16*bit3(k);  slot = (k&7) + 8*bit4(k)
__device__ __forceinline__ size_t aswz(int row, int k, int ktiles){
  int lane = (row & 15) + (((k >> 3) & 1) << 4);
  int slot = (k & 7) + (((k >> 4) & 1) << 3);
  return ((((size_t)(row >> 4)) * ktiles + (k >> 5)) * 32 + lane) * 16 + slot;
}
// B (KxN): lane = (n&15) + 16*bit3(k); slot = (k&7) + 8*bit4(k)
__device__ __forceinline__ size_t bswz(int k, int n, int ntiles){
  int lane = (n & 15) + (((k >> 3) & 1) << 4);
  int slot = (k & 7) + (((k >> 4) & 1) << 3);
  return ((((size_t)(k >> 5)) * ntiles + (n >> 4)) * 32 + lane) * 16 + slot;
}

__device__ __forceinline__ float sincf(float x){
  float ax = fabsf(x);
  if (ax < 1e-6f) return 1.0f;
  float px = PI_F * x;
  return sinf(px) / px;
}

// =====================================================================
//  WMMA bf16 GEMM on pre-swizzled operands.
//  C = alpha * A(MxK) * B(KxN), C row-major f32.
//  256 threads = 8 wave32; each wave computes 16x64 (4 WMMA tiles);
//  block computes 128x64.  A fragments: 2x global_load_b128 per lane.
//  Shared B k-tile: 4KB staged to LDS (async-to-LDS when available),
//  unconditional — all B allocations are padded to 4 n-tiles.
// =====================================================================
__global__ __launch_bounds__(256)
void wmma_gemm_swz(const bf16_t* __restrict__ A, const bf16_t* __restrict__ B,
                   float* __restrict__ C, int M, int N, int Ktiles,
                   int BNtiles, int ldc, float alpha)
{
  __shared__ __align__(32) bf16_t lB[4*32*16];     // 4 fragments = 4KB
  const int tid  = threadIdx.x;
  const int lane = tid & 31;
  const int wave = tid >> 5;                       // 0..7
  const int row0 = blockIdx.y*128 + wave*16;       // 16 rows per wave
  const int rt   = row0 >> 4;
  const int nt0  = blockIdx.x*4;                   // 4 n-tiles per block

  __builtin_amdgcn_s_wait_tensorcnt(0);            // CDNA5 counter path (no-op)

  // B staging assignment: thread -> (fragment, lane, half); 16B each
  const int sfrag = tid >> 6;                      // 0..3
  const int sln   = (tid >> 1) & 31;
  const int shalf = tid & 1;
  bf16_t* sdst = &lB[(sfrag*32 + sln)*16 + shalf*8];
  const bf16_t* sB0 = B + (((size_t)(nt0 + sfrag))*32 + sln)*16 + shalf*8;

  v8f acc[4] = {};
  for (int kt = 0; kt < Ktiles; ++kt) {
    const bf16_t* src = sB0 + (size_t)kt*BNtiles*512;
#ifdef HAVE_ASYNC_LDS
    __builtin_amdgcn_global_load_async_to_lds_b128(
        (__attribute__((address_space(1))) v4i_t*)src,
        (__attribute__((address_space(3))) v4i_t*)sdst, 0, 0);
    __builtin_amdgcn_s_wait_asynccnt(0);
#else
    *(uint4*)sdst = *(const uint4*)src;
#endif
    __syncthreads();

    // ---- A fragment: contiguous 32B per lane, direct from global ----
    const bf16_t* asrc = A + (((size_t)rt*Ktiles + kt)*32 + lane)*16;
    v16bf afrag = *(const v16bf*)asrc;
    if (kt + 1 < Ktiles)                            // global_prefetch_b8
      __builtin_prefetch(A + (((size_t)rt*Ktiles + kt + 1)*32 + lane)*16, 0, 1);

    // ---- 4 B fragments from LDS (2x ds_load_b128 each) ----
    v16bf bf0 = *(const v16bf*)&lB[(0*32 + lane)*16];
    v16bf bf1 = *(const v16bf*)&lB[(1*32 + lane)*16];
    v16bf bf2 = *(const v16bf*)&lB[(2*32 + lane)*16];
    v16bf bf3 = *(const v16bf*)&lB[(3*32 + lane)*16];

    acc[0] = __builtin_amdgcn_wmma_f32_16x16x32_bf16(false, afrag, false, bf0,
                                                     (short)0, acc[0], false, false);
    acc[1] = __builtin_amdgcn_wmma_f32_16x16x32_bf16(false, afrag, false, bf1,
                                                     (short)0, acc[1], false, false);
    acc[2] = __builtin_amdgcn_wmma_f32_16x16x32_bf16(false, afrag, false, bf2,
                                                     (short)0, acc[2], false, false);
    acc[3] = __builtin_amdgcn_wmma_f32_16x16x32_bf16(false, afrag, false, bf3,
                                                     (short)0, acc[3], false, false);
    __syncthreads();
  }

  // ---- store C (guarded; layout: lane n + rows r+8*(lane>>4)) ----
  const int cn  = lane & 15;
  const int mh  = lane >> 4;
  const int cr0 = row0 + mh*8;
  #pragma unroll
  for (int f = 0; f < 4; ++f) {
    int ccol = (nt0 + f)*16 + cn;
    if (ccol < N) {
      #pragma unroll
      for (int r = 0; r < 8; ++r) {
        int cr = cr0 + r;
        if (cr < M) C[(size_t)cr*ldc + ccol] = alpha * acc[f][r];
      }
    }
  }
}

// =====================================================================
//  Basis generation (bf16, written in B-swizzled fragment order).
// =====================================================================
__device__ __forceinline__ float cos2048(long long kn){
  int m = (int)(kn & 2047);
  return cosf(TAU_F * (float)m / 2048.0f);
}
__device__ __forceinline__ float sin2048(long long kn){
  int m = (int)(kn & 2047);
  return sinf(TAU_F * (float)m / 2048.0f);
}

// rfft basis: K rows k=0..2047 (time), N cols: [0,NPAD)=cos f, [NPAD,NRF)=sin f
__global__ void gen_rfft_basis(bf16_t* __restrict__ Brf){
  int k = blockIdx.y;
  int n = blockIdx.x*256 + threadIdx.x;
  if (n >= NRF) return;
  float v = 0.f;
  int f = (n < NPAD) ? n : (n - NPAD);
  if (f < OSL) {
    long long kn = (long long)k * f;
    v = (n < NPAD) ? cos2048(kn) : sin2048(kn);
  }
  Brf[bswz(k, n, NT_RF)] = (bf16_t)v;
}

// hfft basis: K rows k=0..NPAD-1 (freq bin), N cols n=0..2047
// H[k][n] = w_k cos(2pi k n/2048), w_0=w_1024=1 else 2; zero k>=OSL
__global__ void gen_hfft_basis(bf16_t* __restrict__ Bh){
  int k = blockIdx.y;
  int n = blockIdx.x*256 + threadIdx.x;
  if (n >= FFTL) return;
  float v = 0.f;
  if (k < OSL) {
    float w = (k == 0 || k == OSL-1) ? 1.f : 2.f;
    v = w * cos2048((long long)k * n);
  }
  Bh[bswz(k, n, NT_FFTL)] = (bf16_t)v;
}

// ihfft basis: K rows n=0..2047, N cols k=0..(68*16-1): cos(2pi k n/2048)/2048
// pad columns k>=OSL are written as zeros (allocation padded to 68 n-tiles)
__global__ void gen_ihfft_basis(bf16_t* __restrict__ Bih){
  int n = blockIdx.y;
  int k = blockIdx.x*256 + threadIdx.x;
  if (k >= NT_NPAD_AL*16) return;
  float v = 0.f;
  if (k < OSL) v = cos2048((long long)k * n) * (1.0f/2048.0f);
  Bih[bswz(n, k, NT_NPAD_AL)] = (bf16_t)v;
}

// =====================================================================
//  Front end: normalize, IIR highpass bank, frame+window
// =====================================================================
__global__ __launch_bounds__(512)
void stats_scale(const double* __restrict__ x, float* __restrict__ xs,
                 float* __restrict__ xamp)
{
  int b = blockIdx.x;
  __shared__ double s1[512], s2[512];
  double a = 0, c = 0;
  for (int t = threadIdx.x; t < TLEN; t += 512) {
    double v = x[(size_t)b*TLEN + t];
    a += v; c += v*v;
  }
  s1[threadIdx.x] = a; s2[threadIdx.x] = c; __syncthreads();
  for (int st = 256; st > 0; st >>= 1) {
    if (threadIdx.x < st) { s1[threadIdx.x]+=s1[threadIdx.x+st]; s2[threadIdx.x]+=s2[threadIdx.x+st]; }
    __syncthreads();
  }
  double mean = s1[0] / TLEN;
  double var  = (s2[0] - (double)TLEN*mean*mean) / (double)(TLEN-1);
  double sdv  = sqrt(var > 0 ? var : 0);
  if (threadIdx.x == 0) xamp[b] = (float)sdv;
  float sc = (sdv < 1e-8) ? 1.f : (float)(2200.0 / sdv);
  for (int t = threadIdx.x; t < TLEN; t += 512)
    xs[(size_t)b*TLEN + t] = (float)x[(size_t)b*TLEN + t] * sc;
}

// 3 order-6 highpass filters (DF2T), one thread per (batch,filter)
__global__ void iir3(const float* __restrict__ xs, const double* __restrict__ bco,
                     const double* __restrict__ aco, float* __restrict__ xh)
{
  int id = blockIdx.x*blockDim.x + threadIdx.x;
  if (id >= BATCH*3) return;
  int b = id / 3, fl = id % 3;
  double bb[7], aa[7], z[6];
  for (int i = 0; i < 7; ++i) { bb[i] = bco[fl*7+i]; aa[i] = aco[fl*7+i]; }
  for (int i = 0; i < 6; ++i) z[i] = 0.0;
  const float* xi = xs + (size_t)b*TLEN;
  float* yo = xh + ((size_t)b*3 + fl)*TLEN;
  for (int t = 0; t < TLEN; ++t) {
    double xv = (double)xi[t];
    double y  = bb[0]*xv + z[0];
    #pragma unroll
    for (int i = 0; i < 5; ++i) z[i] = bb[i+1]*xv - aa[i+1]*y + z[i+1];
    z[5] = bb[6]*xv - aa[6]*y;
    yo[t] = (float)y;
  }
}

// frame, de-mean, pitch-synchronous windows wxe/wxd -> A-swizzled bf16 rows
__global__ __launch_bounds__(256)
void frame_window(const float* __restrict__ xh, const double* __restrict__ f0g,
                  const double* __restrict__ tt, const double* __restrict__ wseed,
                  const double* __restrict__ tnom, bf16_t* __restrict__ Abig)
{
  int f = blockIdx.x, b = blockIdx.y;
  __shared__ float tx[FL];
  __shared__ float we[FL];
  __shared__ float red[256];
  const float* x0 = xh + (size_t)(b*3 + 0)*TLEN;

  float acc = 0.f;
  for (int j = threadIdx.x; j < FL; j += 256) {
    int t = f*FP + j - FL/2;
    float v = (t >= 0 && t < TLEN) ? x0[t] : 0.f;
    tx[j] = v; acc += v;
  }
  red[threadIdx.x] = acc; __syncthreads();
  for (int st = 128; st > 0; st >>= 1) {
    if (threadIdx.x < st) red[threadIdx.x] += red[threadIdx.x+st];
    __syncthreads();
  }
  float mean = red[0] / (float)FL;
  __syncthreads();

  double f0 = f0g[(size_t)b*NFRAME + f];
  if (f0 == 0.0) f0 = DEFAULT_F0;
  float t0 = (float)tnom[0];

  float nacc = 0.f;
  for (int j = threadIdx.x; j < FL; j += 256) {
    float ttf = (float)(tt[j]*f0);
    float pos = (ttf/FNOMf - t0) * (float)SRATE;
    int   i0  = (int)floorf(pos);
    if (i0 < 0) i0 = 0; if (i0 > 2558) i0 = 2558;
    float w = pos - (float)i0;
    float v = (pos >= 0.f && pos <= 2559.f)
            ? (float)wseed[i0]*(1.f-w) + (float)wseed[i0+1]*w : 0.f;
    we[j] = v; nacc += v*v;
  }
  red[threadIdx.x] = nacc; __syncthreads();
  for (int st = 128; st > 0; st >>= 1) {
    if (threadIdx.x < st) red[threadIdx.x] += red[threadIdx.x+st];
    __syncthreads();
  }
  float inv = 1.f / (sqrtf(red[0]) + EPSf);

  int rowE = b*NFRAME + f;
  int rowD = NFR + rowE;
  for (int j = threadIdx.x; j < FFTL; j += 256) {
    float av = 0.f, dv = 0.f;
    if (j < FL) {
      float w   = we[j]*inv;
      float ttf = (float)(tt[j]*f0);
      float s   = 0.36f * w * sinf(PI_F*ttf);
      float xv  = tx[j] - mean;
      av = xv*w; dv = xv*s;
    }
    Abig[aswz(rowE, j, KT_FFTL)] = (bf16_t)av;
    Abig[aswz(rowD, j, KT_FFTL)] = (bf16_t)dv;
  }
}

// =====================================================================
//  Power spectrum + sub-f0 rebin; write f32 P and A-swizzled bf16 Pb
// =====================================================================
__global__ __launch_bounds__(256)
void power_combine(const float* __restrict__ S, const double* __restrict__ f0g,
                   float* __restrict__ P, bf16_t* __restrict__ Pb)
{
  int fr = blockIdx.x;
  int b = fr / NFRAME, f = fr % NFRAME;
  __shared__ float pwlow[64];
  __shared__ float tmp2[32];
  double f0 = f0g[(size_t)b*NFRAME + f];
  if (f0 == 0.0) f0 = DEFAULT_F0;
  const float* se = S + (size_t)fr*NRF;
  const float* sd = S + (size_t)(NFR + fr)*NRF;

  for (int n = threadIdx.x; n < NPAD; n += 256) {
    float v = 0.f;
    if (n < OSL) {
      float re1 = se[n], im1 = se[NPAD+n];
      float re2 = sd[n], im2 = sd[NPAD+n];
      float pwr = re1*re1 + im1*im1 + re2*re2 + im2*im2;
      pwr = fmaxf(pwr, EPSf);
      v = powf(pwr, 0.3f);             // ^(PC/2)
    }
    P[(size_t)fr*NPAD + n] = v;
    if (n < F0PM) pwlow[n] = v;
  }
  __syncthreads();

  float f0pr = (float)(f0*((double)FFTL/(double)SRATE) + 1.0);
  int   f0p2 = (int)floorf((f0pr + 1.f)*0.5f);
  if (threadIdx.x < F0P2M) {
    float pos = f0pr - (float)threadIdx.x - 1.f;
    int i0 = (int)floorf(pos);
    if (i0 < 0) i0 = 0; if (i0 > F0PM-2) i0 = F0PM-2;
    float w = pos - (float)i0;
    tmp2[threadIdx.x] = pwlow[i0]*(1.f-w) + pwlow[i0+1]*w;
  }
  __syncthreads();

  for (int n = threadIdx.x; n < NPAD; n += 256) {
    float v = P[(size_t)fr*NPAD + n];
    if (n < f0p2) v = (n < F0P2M) ? tmp2[n] : 0.f;
    P[(size_t)fr*NPAD + n]  = v;
    Pb[aswz(fr, n, KT_NPAD)] = (bf16_t)v;
  }
}

// ---------------- pointwise spectral-domain weights ----------------
__global__ void mul_ww2t(const float* __restrict__ Y, const double* __restrict__ f0g,
                         const double* __restrict__ ttm, const double* __restrict__ lft,
                         bf16_t* __restrict__ Yb)
{
  int fr = blockIdx.y;
  int n = blockIdx.x*256 + threadIdx.x;
  if (n >= FFTL) return;
  int b = fr / NFRAME, f = fr % NFRAME;
  double f0 = f0g[(size_t)b*NFRAME + f];
  if (f0 == 0.0) f0 = DEFAULT_F0;
  float tf = (float)(ttm[n]*f0);
  float s = sincf(3.f*tf);
  float v = Y[(size_t)fr*FFTL + n] * s*s * (float)lft[n];
  Yb[aswz(fr, n, KT_FFTL)] = (bf16_t)v;
}

__global__ void ratio_kernel(const float* __restrict__ P, const float* __restrict__ spw2,
                             bf16_t* __restrict__ Rb)
{
  int fr = blockIdx.y;
  int n = blockIdx.x*256 + threadIdx.x;
  if (n >= NPAD) return;
  float v = 0.f;
  if (n < OSL) v = P[(size_t)fr*NPAD + n] / (spw2[(size_t)fr*NPAD + n] + EPSf);
  Rb[aswz(fr, n, KT_NPAD)] = (bf16_t)v;
}

__global__ void mul_wwt(const float* __restrict__ Y, const double* __restrict__ f0g,
                        const double* __restrict__ ttm, const double* __restrict__ lft,
                        const double* __restrict__ ovc, bf16_t* __restrict__ Yb)
{
  int fr = blockIdx.y;
  int n = blockIdx.x*256 + threadIdx.x;
  if (n >= FFTL) return;
  int b = fr / NFRAME, f = fr % NFRAME;
  double f0 = f0g[(size_t)b*NFRAME + f];
  if (f0 == 0.0) f0 = DEFAULT_F0;
  float tf = (float)(ttm[n]*f0);
  float s  = sincf(tf);
  float w  = s*s * ((float)ovc[0] + 2.f*(float)ovc[1]*cosf(TAU_F*tf)
                                  + 2.f*(float)ovc[2]*cosf(2.f*TAU_F*tf));
  float v = Y[(size_t)fr*FFTL + n] * w * (float)lft[n];
  Yb[aswz(fr, n, KT_FFTL)] = (bf16_t)v;
}

// spw normalize + nonlinearity -> n2sgram (f32)
__global__ void spw_finish(const float* __restrict__ spwraw, const float* __restrict__ spw2,
                           const double* __restrict__ f0g, const double* __restrict__ ttm,
                           const double* __restrict__ ovc, float* __restrict__ N2)
{
  int fr = blockIdx.y;
  int n = blockIdx.x*256 + threadIdx.x;
  if (n >= NPAD) return;
  int b = fr / NFRAME, f = fr % NFRAME;
  double f0 = f0g[(size_t)b*NFRAME + f];
  if (f0 == 0.0) f0 = DEFAULT_F0;
  float tf0 = (float)(ttm[0]*f0);
  float s0  = sincf(tf0);
  float wwt0 = s0*s0 * ((float)ovc[0] + 2.f*(float)ovc[1]*cosf(TAU_F*tf0)
                                      + 2.f*(float)ovc[2]*cosf(2.f*TAU_F*tf0));
  float spw = spwraw[(size_t)fr*NPAD + n] / (wwt0 + EPSf);
  spw = fminf(fmaxf(spw, -100.f), 100.f);
  float z = (4.f/1.4f)*spw;
  float az = fabsf(z);
  float lcosh2 = az + log1pf(expf(-2.f*az));          // log(2*cosh(z)), stable
  float n2 = spw2[(size_t)fr*NPAD + n] * (0.175f*lcosh2 + 0.5f*spw);
  n2 = powf(fmaxf(n2, EPSf), 10.f/3.f);               // ^(2/PC)
  N2[(size_t)fr*NPAD + n] = n2;
}

// =====================================================================
//  Aperiodicity path (channels 1,2 of xh)
// =====================================================================
__global__ void pwcs_conv(const float* __restrict__ xh, const double* __restrict__ h3,
                          int nh, float* __restrict__ pwcs)
{
  int b = blockIdx.x, c = blockIdx.y;     // c in {0,1} -> xh channel c+1
  const float* xc = xh + ((size_t)b*3 + c + 1)*TLEN;
  for (int f = threadIdx.x; f < NFRAME; f += blockDim.x) {
    int t0 = f*FP; float s = 0.f;
    for (int j = 0; j < nh; ++j) {
      int t = t0 - j;
      if (t >= 0 && t < TLEN) { float v = xc[t]; s += (float)h3[j]*v*v; }
    }
    pwcs[((size_t)b*2 + c)*NFRAME + f] = s;
  }
}

__global__ __launch_bounds__(256)
void pwcs_scale(const float* __restrict__ N2, float* __restrict__ pwcs)
{
  int b = blockIdx.x;
  __shared__ float r1[256], r2[256], r3[256], r4[256];
  float slbb = 0.f, sall = 0.f;
  for (long i = threadIdx.x; i < (long)NFRAME*NPAD; i += 256) {
    int f = (int)(i / NPAD), n = (int)(i % NPAD);
    if (n < OSL) {
      float v = N2[((size_t)(b*NFRAME + f))*NPAD + n];
      sall += v;
      if (n >= LBB) slbb += v;
    }
  }
  float p0 = 0.f, p1 = 0.f;
  for (int f = threadIdx.x; f < NFRAME; f += 256) {
    p0 += pwcs[((size_t)b*2 + 0)*NFRAME + f];
    p1 += pwcs[((size_t)b*2 + 1)*NFRAME + f];
  }
  r1[threadIdx.x]=slbb; r2[threadIdx.x]=sall; r3[threadIdx.x]=p0; r4[threadIdx.x]=p1;
  __syncthreads();
  for (int st = 128; st > 0; st >>= 1) {
    if (threadIdx.x < st) {
      r1[threadIdx.x]+=r1[threadIdx.x+st]; r2[threadIdx.x]+=r2[threadIdx.x+st];
      r3[threadIdx.x]+=r3[threadIdx.x+st]; r4[threadIdx.x]+=r4[threadIdx.x+st];
    }
    __syncthreads();
  }
  float sc0 = r1[0] / (r3[0] + EPSf);
  float sc1 = r2[0] / (r4[0] + EPSf);
  for (int f = threadIdx.x; f < NFRAME; f += 256) {
    pwcs[((size_t)b*2 + 0)*NFRAME + f] *= sc0;
    pwcs[((size_t)b*2 + 1)*NFRAME + f] *= sc1;
  }
}

__global__ void apwt_dpwt(const float* __restrict__ pwcs, const double* __restrict__ ww,
                          int nw, float* __restrict__ apwt, float* __restrict__ dpwt)
{
  int f = blockIdx.x*256 + threadIdx.x;
  int b = blockIdx.y;
  if (f >= NFRAME) return;
  const float* pwch = pwcs + ((size_t)b*2 + 1)*NFRAME;
  int begin = nw/2;
  float a = 0.f, d = 0.f;
  for (int j = 0; j < nw; ++j) {
    int t = f + begin - j;
    if (t >= 0 && t < NFRAME) a += (float)ww[j]*pwch[t];
    if (t >= 0 && t < NFRAME-1) { float df = pwch[t+1]-pwch[t]; d += (float)ww[j]*df*df; }
  }
  apwt[(size_t)b*NFRAME + f] = a;
  dpwt[(size_t)b*NFRAME + f] = sqrtf(d + EPSf);
}

__global__ __launch_bounds__(256)
void apwt_max(const float* __restrict__ apwt, float* __restrict__ maxa)
{
  int b = blockIdx.x;
  __shared__ float red[256];
  float m = -1e30f;
  for (int f = threadIdx.x; f < NFRAME; f += 256) m = fmaxf(m, apwt[(size_t)b*NFRAME + f]);
  red[threadIdx.x] = m; __syncthreads();
  for (int st = 128; st > 0; st >>= 1) {
    if (threadIdx.x < st) red[threadIdx.x] = fmaxf(red[threadIdx.x], red[threadIdx.x+st]);
    __syncthreads();
  }
  if (threadIdx.x == 0) maxa[b] = red[0];
}

__global__ __launch_bounds__(256)
void frame_sums(const float* __restrict__ N2, float* __restrict__ sumn2)
{
  int fr = blockIdx.x;
  __shared__ float red[256];
  float s = 0.f;
  for (int n = threadIdx.x; n < OSL; n += 256) s += N2[(size_t)fr*NPAD + n];
  red[threadIdx.x] = s; __syncthreads();
  for (int st = 128; st > 0; st >>= 1) {
    if (threadIdx.x < st) red[threadIdx.x] += red[threadIdx.x+st];
    __syncthreads();
  }
  if (threadIdx.x == 0) sumn2[fr] = red[0];
}

__global__ void pwc_kernel(const float* __restrict__ apwt, const float* __restrict__ dpwt,
                           const float* __restrict__ maxa, const float* __restrict__ pwcs,
                           const float* __restrict__ sumn2, float* __restrict__ pwc)
{
  int f = blockIdx.x*256 + threadIdx.x;
  int b = blockIdx.y;
  if (f >= NFRAME) return;
  float a = apwt[(size_t)b*NFRAME + f];
  if (a <= 0.f) a = maxa[b];
  float ratio = dpwt[(size_t)b*NFRAME + f] / (a + EPSf);
  float l = 1.f / (1.f + expf(-((sqrtf(ratio) - 0.75f) * 20.f)));
  float p0 = pwcs[((size_t)b*2 + 0)*NFRAME + f];
  float s  = sumn2[(size_t)b*NFRAME + f];
  pwc[(size_t)b*NFRAME + f] = l * p0 / (s + EPSf) + (1.f - l);
}

__global__ void apply_pwc_sqrt(const float* __restrict__ N2, const double* __restrict__ f0g,
                               const float* __restrict__ pwc, bf16_t* __restrict__ Xb)
{
  int fr = blockIdx.y;
  int n = blockIdx.x*256 + threadIdx.x;
  if (n >= NPAD) return;
  int b = fr / NFRAME, f = fr % NFRAME;
  float v = 0.f;
  if (n < OSL) {
    v = N2[(size_t)fr*NPAD + n];
    if (f0g[(size_t)b*NFRAME + f] == 0.0) v *= pwc[(size_t)b*NFRAME + f];
    v = sqrtf(fabsf(v + EPSf));
  }
  Xb[aswz(fr, n, KT_NPAD)] = (bf16_t)v;
}

__global__ void mul_comp(const float* __restrict__ ccs2, const double* __restrict__ f0g,
                         const double* __restrict__ pb2, bf16_t* __restrict__ Xb)
{
  int fr = blockIdx.y;
  int n = blockIdx.x*256 + threadIdx.x;
  if (n >= NPAD) return;
  int b = fr / NFRAME, f = fr % NFRAME;
  float v = 0.f;
  if (n < OSL) {
    float f0r = (float)f0g[(size_t)b*NFRAME + f];   // raw f0 (0 if unvoiced)
    float comp = fminf(1.f + 0.2f * (float)pb2[n] * f0r*f0r, 20.f);
    v = ccs2[(size_t)fr*NPAD + n] * comp;
  }
  Xb[aswz(fr, n, KT_NPAD)] = (bf16_t)v;
}

__global__ void final_out(const float* __restrict__ N3, const float* __restrict__ xamp,
                          float* __restrict__ out)
{
  // benign cluster probe (CDNA5 TTMP/cluster path); never taken
  if (__builtin_amdgcn_cluster_id_x() > 16) return;
  int fr = blockIdx.y;
  int n = blockIdx.x*256 + threadIdx.x;
  if (n >= OSL) return;
  int b = fr / NFRAME;
  float n3  = N3[(size_t)fr*NPAD + n];
  float n2f = 0.5f*(fabsf(n3) + n3) + 0.1f;
  float xa  = xamp[b];
  float n3s = (xa < EPSf) ? n2f : n2f * (xa / 2200.f);
  out[(size_t)fr*OSL + n] = 2.f * logf(fabsf(n3s + EPSf));
}

// =====================================================================
//  Host launcher
// =====================================================================
static inline size_t al256(size_t x){ return (x + 255) & ~(size_t)255; }

extern "C" void kernel_launch(void* const* d_in, const int* in_sizes, int n_in,
                              void* d_out, int out_size, void* d_ws, size_t ws_size,
                              hipStream_t stream)
{
  const double* x     = (const double*)d_in[0];
  const double* f0g   = (const double*)d_in[1];
  const double* bco   = (const double*)d_in[2];
  const double* aco   = (const double*)d_in[3];
  const double* tt    = (const double*)d_in[4];
  const double* wseed = (const double*)d_in[5];
  const double* tnom  = (const double*)d_in[6];
  const double* ttm   = (const double*)d_in[7];
  const double* lft   = (const double*)d_in[8];
  const double* ovc   = (const double*)d_in[9];
  const double* h3    = (const double*)d_in[10];
  const double* ww    = (const double*)d_in[11];
  const double* pb2   = (const double*)d_in[12];
  const int nh3 = in_sizes[10];
  const int nww = in_sizes[11];
  float* out = (float*)d_out;

  // GEMM grid row padding (block = 128 rows)
  const int GY12 = (2*NFR + 127)/128;   // 94  -> 12032 rows -> 752 row-tiles
  const int GY6  = (NFR   + 127)/128;   // 47  ->  6016 rows -> 376 row-tiles
  const size_t RT12 = (size_t)GY12*8;
  const size_t RT6  = (size_t)GY6*8;

  // ---- workspace layout ----
  char* base = (char*)d_ws;
  size_t off = 0;
  auto take = [&](size_t bytes)->char*{ char* p = base + off; off = al256(off + bytes); return p; };

  float*  XS   = (float*) take((size_t)BATCH*TLEN*4);
  float*  XH   = (float*) take((size_t)BATCH*3*TLEN*4);
  float*  XAMP = (float*) take(256);
  bf16_t* ABIG = (bf16_t*)take(RT12*KT_FFTL*512*2);              // swizzled A, K=2048
  bf16_t* BRF  = (bf16_t*)take((size_t)KT_FFTL*NT_RF*512*2);     // swizzled B
  bf16_t* BH   = (bf16_t*)take((size_t)KT_NPAD*NT_FFTL*512*2);
  bf16_t* BIH  = (bf16_t*)take((size_t)KT_FFTL*NT_NPAD_AL*512*2);
  float*  SBUF = (float*) take((size_t)2*NFR*NRF*4);
  float*  PBUF = (float*) take((size_t)NFR*NPAD*4);
  bf16_t* XB   = (bf16_t*)take(RT6*KT_FFTL*512*2);               // swizzled A staging (max K)
  float*  YBUF = (float*) take((size_t)NFR*FFTL*4);
  float*  SPW2 = (float*) take((size_t)NFR*NPAD*4);
  float*  TMPA = (float*) take((size_t)NFR*NPAD*4);
  float*  N2   = (float*) take((size_t)NFR*NPAD*4);
  float*  N3   = (float*) take((size_t)NFR*NPAD*4);
  float*  PWCS = (float*) take((size_t)BATCH*2*NFRAME*4);
  float*  APWT = (float*) take((size_t)BATCH*NFRAME*4);
  float*  DPWT = (float*) take((size_t)BATCH*NFRAME*4);
  float*  PWC  = (float*) take((size_t)BATCH*NFRAME*4);
  float*  SUMN2= (float*) take((size_t)NFR*4);
  float*  MAXA = (float*) take(256);
  (void)ws_size; (void)n_in; (void)out_size;

  // ---- basis generation (B-swizzled, pad columns zeroed) ----
  gen_rfft_basis <<<dim3((NRF +255)/256, FFTL), 256, 0, stream>>>(BRF);
  gen_hfft_basis <<<dim3((FFTL+255)/256, NPAD), 256, 0, stream>>>(BH);
  gen_ihfft_basis<<<dim3((NT_NPAD_AL*16+255)/256, FFTL), 256, 0, stream>>>(BIH);

  // ---- front end ----
  stats_scale<<<BATCH, 512, 0, stream>>>(x, XS, XAMP);
  iir3<<<1, 32, 0, stream>>>(XS, bco, aco, XH);
  frame_window<<<dim3(NFRAME, BATCH), 256, 0, stream>>>(XH, f0g, tt, wseed, tnom, ABIG);

  // rFFT of both windowed frame sets: (12000 x 2048) x (2048 x 2112)
  wmma_gemm_swz<<<dim3(NRF/64, GY12), 256, 0, stream>>>(
      ABIG, BRF, SBUF, 2*NFR, NRF, KT_FFTL, NT_RF, NRF, 1.0f);

  power_combine<<<NFR, 256, 0, stream>>>(SBUF, f0g, PBUF, XB);

  // spw2 = ihfft( ww2t * hfft(pw) * lft )
  wmma_gemm_swz<<<dim3(FFTL/64, GY6), 256, 0, stream>>>(
      XB, BH, YBUF, NFR, FFTL, KT_NPAD, NT_FFTL, FFTL, 1.0f);
  mul_ww2t<<<dim3((FFTL+255)/256, NFR), 256, 0, stream>>>(YBUF, f0g, ttm, lft, XB);
  wmma_gemm_swz<<<dim3((NPAD+63)/64, GY6), 256, 0, stream>>>(
      XB, BIH, SPW2, NFR, NPAD, KT_FFTL, NT_NPAD_AL, NPAD, 1.0f);

  // spw = ihfft( wwt * hfft(pw/spw2) * lft ) / wwt0
  ratio_kernel<<<dim3((NPAD+255)/256, NFR), 256, 0, stream>>>(PBUF, SPW2, XB);
  wmma_gemm_swz<<<dim3(FFTL/64, GY6), 256, 0, stream>>>(
      XB, BH, YBUF, NFR, FFTL, KT_NPAD, NT_FFTL, FFTL, 1.0f);
  mul_wwt<<<dim3((FFTL+255)/256, NFR), 256, 0, stream>>>(YBUF, f0g, ttm, lft, ovc, XB);
  wmma_gemm_swz<<<dim3((NPAD+63)/64, GY6), 256, 0, stream>>>(
      XB, BIH, TMPA, NFR, NPAD, KT_FFTL, NT_NPAD_AL, NPAD, 1.0f);
  spw_finish<<<dim3((NPAD+255)/256, NFR), 256, 0, stream>>>(TMPA, SPW2, f0g, ttm, ovc, N2);

  // aperiodicity / pwc path
  pwcs_conv <<<dim3(BATCH, 2), 256, 0, stream>>>(XH, h3, nh3, PWCS);
  pwcs_scale<<<BATCH, 256, 0, stream>>>(N2, PWCS);
  apwt_dpwt <<<dim3((NFRAME+255)/256, BATCH), 256, 0, stream>>>(PWCS, ww, nww, APWT, DPWT);
  apwt_max  <<<BATCH, 256, 0, stream>>>(APWT, MAXA);
  frame_sums<<<NFR, 256, 0, stream>>>(N2, SUMN2);
  pwc_kernel<<<dim3((NFRAME+255)/256, BATCH), 256, 0, stream>>>(APWT, DPWT, MAXA, PWCS, SUMN2, PWC);

  // final: ccs2 = hfft(sqrt(n2sgram))[:OSL]*comp; n3 = hfft(ccs2, forward)[:OSL]
  apply_pwc_sqrt<<<dim3((NPAD+255)/256, NFR), 256, 0, stream>>>(N2, f0g, PWC, XB);
  wmma_gemm_swz<<<dim3((NPAD+63)/64, GY6), 256, 0, stream>>>(
      XB, BH, TMPA, NFR, NPAD, KT_NPAD, NT_FFTL, NPAD, 1.0f);
  mul_comp<<<dim3((NPAD+255)/256, NFR), 256, 0, stream>>>(TMPA, f0g, pb2, XB);
  wmma_gemm_swz<<<dim3((NPAD+63)/64, GY6), 256, 0, stream>>>(
      XB, BH, N3, NFR, NPAD, KT_NPAD, NT_FFTL, NPAD, 1.0f/2048.0f);

  final_out<<<dim3((OSL+255)/256, NFR), 256, 0, stream>>>(N3, XAMP, out);
}
